// RotaryMultiheadAttention_66769561584091
// MI455X (gfx1250) — matrix operand order repaired and checked by reading, not compiled, not run on hardware
//
#include <hip/hip_runtime.h>
#include <hip/hip_bf16.h>
#include <math.h>

// ---------------------------------------------------------------------------
// RotaryMultiheadAttention for MI455X (gfx1250, wave32, WMMA)
//   B=4, S=2048, D=1024, H=16, dh=64
//   fp32 -> bf16 (staged) -> v_wmma_f32_16x16x32_bf16 -> fp32 accumulate
//   + async global->LDS staging (ASYNCcnt) where the toolchain provides it
// ---------------------------------------------------------------------------

typedef __bf16 bf16_t;
typedef __attribute__((ext_vector_type(16))) __bf16 v16bf;
typedef __attribute__((ext_vector_type(8)))  float  v8f;
typedef int v4i __attribute__((vector_size(4 * sizeof(int))));

#define BATCH 4
#define SEQ   2048
#define DMODEL 1024
#define NHEAD 16
#define DHEAD 64
#define MROWS (BATCH * SEQ)          // 8192
#define LN10000 9.210340371976184f   // ln(10000)

#define AS1 __attribute__((address_space(1)))
#define AS3 __attribute__((address_space(3)))

#if defined(__has_builtin)
#  if __has_builtin(__builtin_amdgcn_global_load_async_to_lds_b128) && \
      __has_builtin(__builtin_amdgcn_s_wait_asynccnt)
#    define HAVE_ASYNC 1
#  endif
#endif
#ifndef HAVE_ASYNC
#  define HAVE_ASYNC 0
#endif

__device__ __forceinline__ v8f wmma_bf16(const v16bf& a, const v16bf& b, const v8f& c) {
    return __builtin_amdgcn_wmma_f32_16x16x32_bf16(false, a, false, b, (short)0, c, false, false);
}
__device__ __forceinline__ v8f zero8() {
    v8f z;
#pragma unroll
    for (int i = 0; i < 8; ++i) z[i] = 0.0f;
    return z;
}

// ===========================================================================
// Kernel 1: fused QKV projection GEMM + RoPE epilogue
//   grid = (D/128, M/128, 3)   z: 0=Q(rope) 1=K(rope) 2=V
//   block = 256 (8 waves), block tile 128x128, wave tile 32x64
//   Register-pipelined double-buffered LDS: 1 barrier / K-step.
//   out layout: [B, H, S, dh] bf16
// ===========================================================================
__global__ __launch_bounds__(256) void qkv_rope_gemm(
    const float* __restrict__ X,
    const float* __restrict__ Wq, const float* __restrict__ bq,
    const float* __restrict__ Wk, const float* __restrict__ bk,
    const float* __restrict__ Wv, const float* __restrict__ bv,
    bf16_t* __restrict__ qo, bf16_t* __restrict__ ko, bf16_t* __restrict__ vo)
{
    __shared__ bf16_t As[2][128][40];   // [buf][m][k] (+pad)
    __shared__ bf16_t Bs[2][128][40];   // [buf][n][k] (+pad)  B[k][n] = W[n][k]

    const int z   = blockIdx.z;
    const float* __restrict__ W   = (z == 0) ? Wq : (z == 1) ? Wk : Wv;
    const float* __restrict__ bia = (z == 0) ? bq : (z == 1) ? bk : bv;
    bf16_t* __restrict__ out      = (z == 0) ? qo : (z == 1) ? ko : vo;
    const bool do_rope = (z < 2);

    const int tid  = threadIdx.x;
    const int wid  = tid >> 5;
    const int lane = tid & 31;
    const int l16  = lane & 15;
    const int lh   = lane >> 4;          // lane half (0/1)
    const int wm   = wid & 3;            // wave M index (4)
    const int wn   = wid >> 2;           // wave N index (2)
    const int m0   = blockIdx.y * 128;
    const int n0   = blockIdx.x * 128;

    v8f acc[2][4];
#pragma unroll
    for (int mi = 0; mi < 2; ++mi)
#pragma unroll
        for (int ni = 0; ni < 4; ++ni) acc[mi][ni] = zero8();

    float4 ra[4], rb[4];
    auto gload = [&](int k0) {
#pragma unroll
        for (int it = 0; it < 4; ++it) {
            const int f  = tid + 256 * it;     // float4 chunk id
            const int r  = f >> 3;             // row 0..127
            const int kq = (f & 7) << 2;       // k offset 0..28
            ra[it] = *reinterpret_cast<const float4*>(X + (size_t)(m0 + r) * DMODEL + k0 + kq);
            rb[it] = *reinterpret_cast<const float4*>(W + (size_t)(n0 + r) * DMODEL + k0 + kq);
        }
    };
    auto lstore = [&](int nb) {
#pragma unroll
        for (int it = 0; it < 4; ++it) {
            const int f  = tid + 256 * it;
            const int r  = f >> 3;
            const int kq = (f & 7) << 2;
            As[nb][r][kq + 0] = (bf16_t)ra[it].x; As[nb][r][kq + 1] = (bf16_t)ra[it].y;
            As[nb][r][kq + 2] = (bf16_t)ra[it].z; As[nb][r][kq + 3] = (bf16_t)ra[it].w;
            Bs[nb][r][kq + 0] = (bf16_t)rb[it].x; Bs[nb][r][kq + 1] = (bf16_t)rb[it].y;
            Bs[nb][r][kq + 2] = (bf16_t)rb[it].z; Bs[nb][r][kq + 3] = (bf16_t)rb[it].w;
        }
    };

    gload(0);
    lstore(0);
    __syncthreads();

    int buf = 0;
    for (int kt = 0; kt < DMODEL / 32; ++kt) {
        const bool more = (kt + 1 < DMODEL / 32);
        if (more) {
            gload((kt + 1) * 32);
            if (kt + 2 < DMODEL / 32) {     // hint 2 tiles ahead -> global_prefetch_b8
                __builtin_prefetch(X + (size_t)(m0 + (tid >> 3)) * DMODEL + (kt + 2) * 32, 0, 1);
                __builtin_prefetch(W + (size_t)(n0 + (tid >> 3)) * DMODEL + (kt + 2) * 32, 0, 1);
            }
        }

        // ---- A fragments (16x32, contiguous k = 16*lh + i) ----
        v16bf a[2];
#pragma unroll
        for (int mi = 0; mi < 2; ++mi) {
            const int row = wm * 32 + mi * 16 + l16;
#pragma unroll
            for (int i = 0; i < 16; i += 2) {
                const int kk = i + 16 * lh;
                a[mi][i]     = As[buf][row][kk];
                a[mi][i + 1] = As[buf][row][kk + 1];
            }
        }
        // ---- B fragments (32x16, contiguous k = 16*lh + i, same as A): 16 contiguous halves ----
        v16bf b[4];
#pragma unroll
        for (int ni = 0; ni < 4; ++ni) {
            const int col = wn * 64 + ni * 16 + l16;
            const bf16_t* src = &Bs[buf][col][16 * lh];
#pragma unroll
            for (int i = 0; i < 16; ++i) b[ni][i] = src[i];
        }
#pragma unroll
        for (int mi = 0; mi < 2; ++mi)
#pragma unroll
            for (int ni = 0; ni < 4; ++ni)
                acc[mi][ni] = wmma_bf16(a[mi], b[ni], acc[mi][ni]);

        if (more) lstore(buf ^ 1);
        __syncthreads();
        buf ^= 1;
    }

    // ---- epilogue: bias, RoPE (even/odd pair via shfl_xor 1), store bf16 ----
#pragma unroll
    for (int mi = 0; mi < 2; ++mi) {
#pragma unroll
        for (int ni = 0; ni < 4; ++ni) {
            const int nn   = n0 + wn * 64 + ni * 16 + l16;   // output column
            const float bv_ = bia[nn];
#pragma unroll
            for (int r = 0; r < 8; ++r) {
                const int mm = m0 + wm * 32 + mi * 16 + r + 8 * lh;  // output row
                float val = acc[mi][ni][r] + bv_;
                if (do_rope) {
                    const float partner = __shfl_xor(val, 1, 32);
                    const int   s   = mm & (SEQ - 1);
                    const float inv = __expf(-(float)(nn & ~1) * (LN10000 / (float)DMODEL));
                    const float ang = (float)s * inv;
                    float sn, cs;
                    __sincosf(ang, &sn, &cs);
                    val = (nn & 1) ? (partner * sn + val * cs)   // odd : x1*sin + x2*cos
                                   : (val * cs - partner * sn);  // even: x1*cos - x2*sin
                }
                const int bi = mm >> 11, s = mm & (SEQ - 1);
                const int h  = nn >> 6,  d = nn & 63;
                out[(((size_t)bi * NHEAD + h) * SEQ + s) * DHEAD + d] = (bf16_t)val;
            }
        }
    }
}

// ===========================================================================
// Kernel 2: flash attention, one block = 128 query rows of one (b,h)
//   grid = (S/128, B*H), block = 256 (8 waves, 16 q-rows each)
//   online softmax over 64 chunks of 32 keys; K/V LDS double-buffered,
//   K chunk staged via GLOBAL_LOAD_ASYNC_TO_LDS (if available); 1 barrier/chunk
// ===========================================================================
__global__ __launch_bounds__(256) void flash_attn(
    const bf16_t* __restrict__ q, const bf16_t* __restrict__ k,
    const bf16_t* __restrict__ v, bf16_t* __restrict__ heads)
{
    __shared__ bf16_t Ks[2][32][72];    // [buf][key][d]   (64 + pad)
    __shared__ bf16_t Vt[2][64][40];    // [buf][d][key]   (32 + pad)
    __shared__ bf16_t Ps[8][16][32];    // per-wave P tile (C->A relayout)

    const int bh   = blockIdx.y;                 // b*H + h
    const int b    = bh >> 4, h = bh & 15;
    const int tid  = threadIdx.x;
    const int wid  = tid >> 5;
    const int lane = tid & 31;
    const int l16  = lane & 15;
    const int lh   = lane >> 4;
    const size_t base = (size_t)bh * SEQ * DHEAD;
    const int q0   = blockIdx.x * 128 + wid * 16;

    // ---- load Q fragments once (A layout, 16 rows x 64 d = 2 frags) ----
    v16bf qa[2];
#pragma unroll
    for (int kk = 0; kk < 2; ++kk) {
        const int row = q0 + l16;
#pragma unroll
        for (int i = 0; i < 16; i += 2) {
            const int d = kk * 32 + i + 16 * lh;
            qa[kk][i]     = q[base + (size_t)row * DHEAD + d];
            qa[kk][i + 1] = q[base + (size_t)row * DHEAD + d + 1];
        }
    }

    v8f o[4];
#pragma unroll
    for (int vi = 0; vi < 4; ++vi) o[vi] = zero8();
    float mrow[8], lrow[8];
#pragma unroll
    for (int r = 0; r < 8; ++r) { mrow[r] = -INFINITY; lrow[r] = 0.0f; }

    // cooperative stage of one 32-key chunk into buffer nb
    auto stage = [&](int kc, int nb) {
        const int f  = tid * 8;          // 8 bf16 per thread
        const int kr = f >> 6;           // key row 0..31
        const int d0 = f & 63;
        const bf16_t* gk = k + base + (size_t)(kc + kr) * DHEAD + d0;
#if HAVE_ASYNC
        __builtin_amdgcn_global_load_async_to_lds_b128(
            (AS1 v4i*)(AS1 const v4i*)(const v4i*)gk,
            (AS3 v4i*)(v4i*)&Ks[nb][kr][d0], 0, 0);
#else
        *reinterpret_cast<uint4*>(&Ks[nb][kr][d0]) =
            *reinterpret_cast<const uint4*>(gk);
#endif
        const uint4 vv = *reinterpret_cast<const uint4*>(
            v + base + (size_t)(kc + kr) * DHEAD + d0);
        const bf16_t* vp = reinterpret_cast<const bf16_t*>(&vv);
#pragma unroll
        for (int j = 0; j < 8; ++j) Vt[nb][d0 + j][kr] = vp[j];  // transpose in LDS
    };

    stage(0, 0);
#if HAVE_ASYNC
    __builtin_amdgcn_s_wait_asynccnt(0);
#endif
    __syncthreads();

    int nb = 0;
    for (int kc = 0; kc < SEQ; kc += 32) {
        if (kc + 32 < SEQ) stage(kc + 32, nb ^ 1);   // overlap with compute

        // ---- scores S = Q @ K^T : 16 x 32, two 16x16 C frags ----
        v8f sc[2];
#pragma unroll
        for (int ni = 0; ni < 2; ++ni) {
            const int key = ni * 16 + l16;
            v16bf kb0, kb1;
#pragma unroll
            for (int i = 0; i < 16; ++i) {
                kb0[i] = Ks[nb][key][16 * lh + i];        // K dim = d 0..31
                kb1[i] = Ks[nb][key][32 + 16 * lh + i];   // K dim = d 32..63
            }
            v8f zc = zero8();
            zc = wmma_bf16(qa[0], kb0, zc);
            zc = wmma_bf16(qa[1], kb1, zc);
            sc[ni] = zc;
        }

        // ---- online softmax: row r lives in VGPR r across one 16-lane half ----
        float pr[2][8];
#pragma unroll
        for (int r = 0; r < 8; ++r) {
            float x0 = sc[0][r] * 0.125f;    // 1/sqrt(64)
            float x1 = sc[1][r] * 0.125f;
            float t  = fmaxf(x0, x1);
            t = fmaxf(t, __shfl_xor(t, 1, 32));
            t = fmaxf(t, __shfl_xor(t, 2, 32));
            t = fmaxf(t, __shfl_xor(t, 4, 32));
            t = fmaxf(t, __shfl_xor(t, 8, 32));
            const float mnew  = fmaxf(mrow[r], t);
            const float p0    = __expf(x0 - mnew);
            const float p1    = __expf(x1 - mnew);
            float srow = p0 + p1;
            srow += __shfl_xor(srow, 1, 32);
            srow += __shfl_xor(srow, 2, 32);
            srow += __shfl_xor(srow, 4, 32);
            srow += __shfl_xor(srow, 8, 32);
            const float scale = __expf(mrow[r] - mnew);
            lrow[r] = lrow[r] * scale + srow;
            mrow[r] = mnew;
#pragma unroll
            for (int vi = 0; vi < 4; ++vi) o[vi][r] *= scale;
            pr[0][r] = p0; pr[1][r] = p1;
        }

        // ---- P: C-frag -> LDS -> A-frag. Wave-private region; LDS ops of one
        //      wave execute in order (ISA 7.3), so no block barrier needed. ----
#pragma unroll
        for (int ni = 0; ni < 2; ++ni)
#pragma unroll
            for (int r = 0; r < 8; ++r)
                Ps[wid][r + 8 * lh][ni * 16 + l16] = (bf16_t)pr[ni][r];

        v16bf pa;
#pragma unroll
        for (int i = 0; i < 16; ++i) {
            const int kk = i + 16 * lh;   // key 0..31
            pa[i] = Ps[wid][l16][kk];
        }

        // ---- O += P(16x32) @ V(32x64): B frags contiguous from Vt ----
#pragma unroll
        for (int vi = 0; vi < 4; ++vi) {
            const int d = vi * 16 + l16;
            v16bf vb;
            const bf16_t* src = &Vt[nb][d][16 * lh];
#pragma unroll
            for (int i = 0; i < 16; ++i) vb[i] = src[i];
            o[vi] = wmma_bf16(pa, vb, o[vi]);
        }

#if HAVE_ASYNC
        __builtin_amdgcn_s_wait_asynccnt(0);   // our async stores to LDS complete
#endif
        __syncthreads();                       // publish next chunk / retire this one
        nb ^= 1;
    }

    // ---- normalize and store heads in [B, S, D] bf16 (un-transpose) ----
#pragma unroll
    for (int vi = 0; vi < 4; ++vi) {
#pragma unroll
        for (int r = 0; r < 8; ++r) {
            const int s = q0 + r + 8 * lh;
            const float val = o[vi][r] / lrow[r];
            heads[((size_t)b * SEQ + s) * DMODEL + h * DHEAD + vi * 16 + l16] = (bf16_t)val;
        }
    }
}

// ===========================================================================
// Kernel 3: output projection  out = heads(bf16) @ Wo^T + bo  (fp32 out)
//   grid = (D/128, M/128), block 256, pipelined like kernel 1
// ===========================================================================
__global__ __launch_bounds__(256) void oproj_gemm(
    const bf16_t* __restrict__ Ain, const float* __restrict__ Wo,
    const float* __restrict__ bo, float* __restrict__ out)
{
    __shared__ bf16_t As[2][128][40];
    __shared__ bf16_t Bs[2][128][40];

    const int tid  = threadIdx.x;
    const int wid  = tid >> 5;
    const int lane = tid & 31;
    const int l16  = lane & 15;
    const int lh   = lane >> 4;
    const int wm   = wid & 3;
    const int wn   = wid >> 2;
    const int m0   = blockIdx.y * 128;
    const int n0   = blockIdx.x * 128;

    v8f acc[2][4];
#pragma unroll
    for (int mi = 0; mi < 2; ++mi)
#pragma unroll
        for (int ni = 0; ni < 4; ++ni) acc[mi][ni] = zero8();

    uint4  rA[2];
    float4 rB[4];
    auto gload = [&](int k0) {
#pragma unroll
        for (int it = 0; it < 2; ++it) {
            const int f   = tid + 256 * it;   // 0..511 (8-half chunks)
            const int r   = f >> 2;
            const int off = (f & 3) << 3;
            rA[it] = *reinterpret_cast<const uint4*>(Ain + (size_t)(m0 + r) * DMODEL + k0 + off);
        }
#pragma unroll
        for (int it = 0; it < 4; ++it) {
            const int f  = tid + 256 * it;
            const int r  = f >> 3;
            const int kq = (f & 7) << 2;
            rB[it] = *reinterpret_cast<const float4*>(Wo + (size_t)(n0 + r) * DMODEL + k0 + kq);
        }
    };
    auto lstore = [&](int nb) {
#pragma unroll
        for (int it = 0; it < 2; ++it) {
            const int f   = tid + 256 * it;
            const int r   = f >> 2;
            const int off = (f & 3) << 3;
            *reinterpret_cast<uint4*>(&As[nb][r][off]) = rA[it];
        }
#pragma unroll
        for (int it = 0; it < 4; ++it) {
            const int f  = tid + 256 * it;
            const int r  = f >> 3;
            const int kq = (f & 7) << 2;
            Bs[nb][r][kq + 0] = (bf16_t)rB[it].x; Bs[nb][r][kq + 1] = (bf16_t)rB[it].y;
            Bs[nb][r][kq + 2] = (bf16_t)rB[it].z; Bs[nb][r][kq + 3] = (bf16_t)rB[it].w;
        }
    };

    gload(0);
    lstore(0);
    __syncthreads();

    int buf = 0;
    for (int kt = 0; kt < DMODEL / 32; ++kt) {
        const bool more = (kt + 1 < DMODEL / 32);
        if (more) {
            gload((kt + 1) * 32);
            if (kt + 2 < DMODEL / 32) {
                __builtin_prefetch(Ain + (size_t)(m0 + (tid >> 2)) * DMODEL + (kt + 2) * 32, 0, 1);
                __builtin_prefetch(Wo  + (size_t)(n0 + (tid >> 3)) * DMODEL + (kt + 2) * 32, 0, 1);
            }
        }

        v16bf a[2];
#pragma unroll
        for (int mi = 0; mi < 2; ++mi) {
            const int row = wm * 32 + mi * 16 + l16;
#pragma unroll
            for (int i = 0; i < 16; i += 2) {
                const int kk = i + 16 * lh;
                a[mi][i]     = As[buf][row][kk];
                a[mi][i + 1] = As[buf][row][kk + 1];
            }
        }
        v16bf bfr[4];
#pragma unroll
        for (int ni = 0; ni < 4; ++ni) {
            const int col = wn * 64 + ni * 16 + l16;
            const bf16_t* src = &Bs[buf][col][16 * lh];
#pragma unroll
            for (int i = 0; i < 16; ++i) bfr[ni][i] = src[i];
        }
#pragma unroll
        for (int mi = 0; mi < 2; ++mi)
#pragma unroll
            for (int ni = 0; ni < 4; ++ni)
                acc[mi][ni] = wmma_bf16(a[mi], bfr[ni], acc[mi][ni]);

        if (more) lstore(buf ^ 1);
        __syncthreads();
        buf ^= 1;
    }

#pragma unroll
    for (int mi = 0; mi < 2; ++mi) {
#pragma unroll
        for (int ni = 0; ni < 4; ++ni) {
            const int nn = n0 + wn * 64 + ni * 16 + l16;
            const float bv_ = bo[nn];
#pragma unroll
            for (int r = 0; r < 8; ++r) {
                const int mm = m0 + wm * 32 + mi * 16 + r + 8 * lh;
                out[(size_t)mm * DMODEL + nn] = acc[mi][ni][r] + bv_;
            }
        }
    }
}

// ===========================================================================
extern "C" void kernel_launch(void* const* d_in, const int* in_sizes, int n_in,
                              void* d_out, int out_size, void* d_ws, size_t ws_size,
                              hipStream_t stream)
{
    (void)in_sizes; (void)n_in; (void)out_size; (void)ws_size;
    const float* X  = (const float*)d_in[0];
    const float* Wq = (const float*)d_in[1];
    const float* bq = (const float*)d_in[2];
    const float* Wk = (const float*)d_in[3];
    const float* bk = (const float*)d_in[4];
    const float* Wv = (const float*)d_in[5];
    const float* bv = (const float*)d_in[6];
    const float* Wo = (const float*)d_in[7];
    const float* bo = (const float*)d_in[8];
    float* out = (float*)d_out;

    // workspace: q, k, v, heads — each [B,H,S,dh] / [B,S,D] bf16 = 16 MiB
    const size_t SLAB = (size_t)MROWS * DMODEL * sizeof(bf16_t);  // 16 MiB
    char* ws = (char*)d_ws;
    bf16_t* qb = (bf16_t*)(ws + 0 * SLAB);
    bf16_t* kb = (bf16_t*)(ws + 1 * SLAB);
    bf16_t* vb = (bf16_t*)(ws + 2 * SLAB);
    bf16_t* hb = (bf16_t*)(ws + 3 * SLAB);

    dim3 g1(DMODEL / 128, MROWS / 128, 3);
    qkv_rope_gemm<<<g1, 256, 0, stream>>>(X, Wq, bq, Wk, bk, Wv, bv, qb, kb, vb);

    dim3 g2(SEQ / 128, BATCH * NHEAD);
    flash_attn<<<g2, 256, 0, stream>>>(qb, kb, vb, hb);

    dim3 g3(DMODEL / 128, MROWS / 128);
    oproj_gemm<<<g3, 256, 0, stream>>>(hb, Wo, bo, out);
}
